// NystromAttention_3058016715013
// MI455X (gfx1250) — matrix operand order repaired and checked by reading, not compiled, not run on hardware
//
#include <hip/hip_runtime.h>

// ---------------------------------------------------------------------------
// Nystrom attention forward for MI455X (gfx1250, wave32, WMMA bf16)
// Wave tile: 32x32 (2x2 WMMA fragments, 4 v_wmma per 32-wide K step)
// ---------------------------------------------------------------------------

#define BATCH   2
#define NTOK    4096
#define DIM     512
#define HEADS   8
#define DHEAD   64
#define LAND    256          // num landmarks (M)
#define LGRP    16           // tokens per landmark (N/M)
#define PITERS  6
#define QKSCALE 0.125f       // DHEAD^-0.5

typedef __attribute__((ext_vector_type(16))) __bf16 v16bf;
typedef __attribute__((ext_vector_type(8)))  float  v8f;

union Frag {
    v16bf v;
    uint4 q[2];
};

static __device__ __forceinline__ unsigned short f2bf(float f) {
    unsigned u = __float_as_uint(f);
    unsigned r = (u + 0x7FFFu + ((u >> 16) & 1u)) >> 16;   // round to nearest even
    return (unsigned short)r;
}
static __device__ __forceinline__ float b2f(unsigned short s) {
    return __uint_as_float(((unsigned)s) << 16);
}

// ---------------------------------------------------------------------------
// f32 -> bf16 conversion
// ---------------------------------------------------------------------------
__global__ void cvt_bf16_kernel(const float* __restrict__ src,
                                unsigned short* __restrict__ dst, long long n) {
    long long i = (long long)blockIdx.x * blockDim.x + threadIdx.x;
    if (i < n) dst[i] = f2bf(src[i]);
}

// ---------------------------------------------------------------------------
// Generic batched GEMM, one wave = one 32x32 f32 tile, K in steps of 32.
//   TRANSB=true : B operand given as [N,K] row-major  (computes A @ B^T)
//   TRANSB=false: B operand given as [K,N] row-major  (LDS tile transpose)
// Outputs optional f32 and/or bf16 C, optional bias, alpha scaling.
// All of M, N must be multiples of 32; K multiple of 32.
// ---------------------------------------------------------------------------
template <bool TRANSB>
__global__ __launch_bounds__(256) void gemm_bf16(
    const unsigned short* __restrict__ A, const unsigned short* __restrict__ B,
    float* __restrict__ Cf, unsigned short* __restrict__ Cbf,
    const float* __restrict__ bias,
    int Mrows, int Ncols, int K, int lda, int ldb, int ldc,
    long long sA, long long sB, long long sC, float alpha)
{
    const int wave = threadIdx.x >> 5;
    const int lane = threadIdx.x & 31;
    const int ntiles = Ncols >> 5;
    const int mtiles = Mrows >> 5;
    long long t = (long long)blockIdx.x * 8 + wave;
    if (t >= (long long)ntiles * mtiles) return;
    const int mt = (int)(t / ntiles);
    const int nt = (int)(t % ntiles);

    A += (long long)blockIdx.z * sA;
    B += (long long)blockIdx.z * sB;

    const int m  = lane & 15;
    const int hi = lane >> 4;

    v8f acc00 = {}, acc01 = {}, acc10 = {}, acc11 = {};
    Frag a0, a1, b0, b1;

    const unsigned short* arow0 = A + (long long)(mt * 32 + m) * lda + hi * 16;
    const unsigned short* arow1 = arow0 + (long long)16 * lda;

    if (TRANSB) {
        const unsigned short* brow0 = B + (long long)(nt * 32 + m) * ldb + hi * 16;
        const unsigned short* brow1 = brow0 + (long long)16 * ldb;
        for (int k0 = 0; k0 < K; k0 += 32) {
            a0.q[0] = *(const uint4*)(arow0 + k0);
            a0.q[1] = *(const uint4*)(arow0 + k0 + 8);
            a1.q[0] = *(const uint4*)(arow1 + k0);
            a1.q[1] = *(const uint4*)(arow1 + k0 + 8);
            b0.q[0] = *(const uint4*)(brow0 + k0);
            b0.q[1] = *(const uint4*)(brow0 + k0 + 8);
            b1.q[0] = *(const uint4*)(brow1 + k0);
            b1.q[1] = *(const uint4*)(brow1 + k0 + 8);
            acc00 = __builtin_amdgcn_wmma_f32_16x16x32_bf16(false, a0.v, false, b0.v, (short)0, acc00, false, false);
            acc01 = __builtin_amdgcn_wmma_f32_16x16x32_bf16(false, a0.v, false, b1.v, (short)0, acc01, false, false);
            acc10 = __builtin_amdgcn_wmma_f32_16x16x32_bf16(false, a1.v, false, b0.v, (short)0, acc10, false, false);
            acc11 = __builtin_amdgcn_wmma_f32_16x16x32_bf16(false, a1.v, false, b1.v, (short)0, acc11, false, false);
        }
    } else {
        __shared__ __align__(16) unsigned short ldsB[8][32 * 40];
        unsigned short* lb = &ldsB[wave][0];
        const unsigned short* bcol = B + nt * 32;
        for (int k0 = 0; k0 < K; k0 += 32) {
            // stage B[k0+lane][nt*32 .. +32) transposed into LDS as [n][k]
            union { uint4 q[4]; unsigned short s[32]; } row;
            const unsigned short* bp = bcol + (long long)(k0 + lane) * ldb;
            row.q[0] = *(const uint4*)(bp);
            row.q[1] = *(const uint4*)(bp + 8);
            row.q[2] = *(const uint4*)(bp + 16);
            row.q[3] = *(const uint4*)(bp + 24);
#pragma unroll
            for (int n = 0; n < 32; ++n) lb[n * 40 + lane] = row.s[n];
            asm volatile("s_wait_dscnt 0" ::: "memory");

            a0.q[0] = *(const uint4*)(arow0 + k0);
            a0.q[1] = *(const uint4*)(arow0 + k0 + 8);
            a1.q[0] = *(const uint4*)(arow1 + k0);
            a1.q[1] = *(const uint4*)(arow1 + k0 + 8);
            b0.q[0] = *(const uint4*)(lb + m * 40 + hi * 16);
            b0.q[1] = *(const uint4*)(lb + m * 40 + hi * 16 + 8);
            b1.q[0] = *(const uint4*)(lb + (m + 16) * 40 + hi * 16);
            b1.q[1] = *(const uint4*)(lb + (m + 16) * 40 + hi * 16 + 8);
            acc00 = __builtin_amdgcn_wmma_f32_16x16x32_bf16(false, a0.v, false, b0.v, (short)0, acc00, false, false);
            acc01 = __builtin_amdgcn_wmma_f32_16x16x32_bf16(false, a0.v, false, b1.v, (short)0, acc01, false, false);
            acc10 = __builtin_amdgcn_wmma_f32_16x16x32_bf16(false, a1.v, false, b0.v, (short)0, acc10, false, false);
            acc11 = __builtin_amdgcn_wmma_f32_16x16x32_bf16(false, a1.v, false, b1.v, (short)0, acc11, false, false);
            asm volatile("s_wait_dscnt 0" ::: "memory");
        }
    }

    // epilogue: subtile (i,j): rows mt*32+i*16+{0..15}, cols nt*32+j*16+{0..15}
    float*          cfp = Cf  ? Cf  + (long long)blockIdx.z * sC : nullptr;
    unsigned short* cbp = Cbf ? Cbf + (long long)blockIdx.z * sC : nullptr;
    const v8f* accs[2][2] = { { &acc00, &acc01 }, { &acc10, &acc11 } };
#pragma unroll
    for (int i = 0; i < 2; ++i) {
#pragma unroll
        for (int j = 0; j < 2; ++j) {
            const v8f& a = *accs[i][j];
            const int col = nt * 32 + j * 16 + m;
            const int r0  = mt * 32 + i * 16 + (hi ? 8 : 0);
            const float bv = bias ? bias[col] : 0.f;
#pragma unroll
            for (int r = 0; r < 8; ++r) {
                float v = a[r] * alpha + bv;
                long long idx = (long long)(r0 + r) * ldc + col;
                if (cfp) cfp[idx] = v;
                if (cbp) cbp[idx] = f2bf(v);
            }
        }
    }
}

// ---------------------------------------------------------------------------
// QKV GEMM: [8192,512] @ [512,1536], 32x32 wave tiles, epilogue scatters into
// per-head q/k/v. q pre-scaled by DHEAD^-0.5; v also written f32 for the conv.
// ---------------------------------------------------------------------------
__global__ __launch_bounds__(256) void gemm_qkv_kernel(
    const unsigned short* __restrict__ A, const unsigned short* __restrict__ B,
    unsigned short* __restrict__ qb, unsigned short* __restrict__ kb,
    unsigned short* __restrict__ vb, float* __restrict__ vf)
{
    const int K = DIM, lda = DIM, ldb = 3 * HEADS * DHEAD;  // 512, 1536
    const int ntiles = (3 * HEADS * DHEAD) / 32;            // 48
    const int mtiles = (BATCH * NTOK) / 32;                 // 256
    const int wave = threadIdx.x >> 5;
    const int lane = threadIdx.x & 31;
    long long t = (long long)blockIdx.x * 8 + wave;
    if (t >= (long long)ntiles * mtiles) return;
    const int mt = (int)(t / ntiles);
    const int nt = (int)(t % ntiles);

    const int m  = lane & 15;
    const int hi = lane >> 4;

    v8f acc00 = {}, acc01 = {}, acc10 = {}, acc11 = {};
    Frag a0, a1, b0, b1;
    const unsigned short* arow0 = A + (long long)(mt * 32 + m) * lda + hi * 16;
    const unsigned short* arow1 = arow0 + (long long)16 * lda;

    __shared__ __align__(16) unsigned short ldsB[8][32 * 40];
    unsigned short* lb = &ldsB[wave][0];
    const unsigned short* bcol = B + nt * 32;
    for (int k0 = 0; k0 < K; k0 += 32) {
        union { uint4 q[4]; unsigned short s[32]; } row;
        const unsigned short* bp = bcol + (long long)(k0 + lane) * ldb;
        row.q[0] = *(const uint4*)(bp);
        row.q[1] = *(const uint4*)(bp + 8);
        row.q[2] = *(const uint4*)(bp + 16);
        row.q[3] = *(const uint4*)(bp + 24);
#pragma unroll
        for (int n = 0; n < 32; ++n) lb[n * 40 + lane] = row.s[n];
        asm volatile("s_wait_dscnt 0" ::: "memory");
        a0.q[0] = *(const uint4*)(arow0 + k0);
        a0.q[1] = *(const uint4*)(arow0 + k0 + 8);
        a1.q[0] = *(const uint4*)(arow1 + k0);
        a1.q[1] = *(const uint4*)(arow1 + k0 + 8);
        b0.q[0] = *(const uint4*)(lb + m * 40 + hi * 16);
        b0.q[1] = *(const uint4*)(lb + m * 40 + hi * 16 + 8);
        b1.q[0] = *(const uint4*)(lb + (m + 16) * 40 + hi * 16);
        b1.q[1] = *(const uint4*)(lb + (m + 16) * 40 + hi * 16 + 8);
        acc00 = __builtin_amdgcn_wmma_f32_16x16x32_bf16(false, a0.v, false, b0.v, (short)0, acc00, false, false);
        acc01 = __builtin_amdgcn_wmma_f32_16x16x32_bf16(false, a0.v, false, b1.v, (short)0, acc01, false, false);
        acc10 = __builtin_amdgcn_wmma_f32_16x16x32_bf16(false, a1.v, false, b0.v, (short)0, acc10, false, false);
        acc11 = __builtin_amdgcn_wmma_f32_16x16x32_bf16(false, a1.v, false, b1.v, (short)0, acc11, false, false);
        asm volatile("s_wait_dscnt 0" ::: "memory");
    }

    const v8f* accs[2][2] = { { &acc00, &acc01 }, { &acc10, &acc11 } };
#pragma unroll
    for (int i = 0; i < 2; ++i) {
#pragma unroll
        for (int j = 0; j < 2; ++j) {
            const v8f& a = *accs[i][j];
            const int col = nt * 32 + j * 16 + m;     // 0..1535
            const int r0  = mt * 32 + i * 16 + (hi ? 8 : 0);
            const int part = col / 512;               // 0=q 1=k 2=v
            const int rem  = col % 512;
            const int h = rem >> 6, d = rem & 63;
#pragma unroll
            for (int r = 0; r < 8; ++r) {
                const int g = r0 + r;                 // global token 0..8191
                const int b = g >> 12, n = g & 4095;
                const long long hidx = ((((long long)b * HEADS + h) * NTOK) + n) * DHEAD + d;
                float v = a[r];
                if (part == 0)      qb[hidx] = f2bf(v * QKSCALE);
                else if (part == 1) kb[hidx] = f2bf(v);
                else { vb[hidx] = f2bf(v); vf[hidx] = v; }
            }
        }
    }
}

// ---------------------------------------------------------------------------
// Landmark means over groups of 16 consecutive tokens.
// ---------------------------------------------------------------------------
__global__ void landmarks_kernel(const unsigned short* __restrict__ q,
                                 const unsigned short* __restrict__ k,
                                 unsigned short* __restrict__ ql,
                                 unsigned short* __restrict__ kl)
{
    int idx = blockIdx.x * blockDim.x + threadIdx.x;   // bh * 256 * 64
    if (idx >= BATCH * HEADS * LAND * DHEAD) return;
    int d  = idx & 63;
    int m  = (idx >> 6) & (LAND - 1);
    int bh = idx >> 14;
    const long long base = ((long long)bh * NTOK + m * LGRP) * DHEAD + d;
    float sq = 0.f, sk = 0.f;
#pragma unroll
    for (int j = 0; j < LGRP; ++j) {
        sq += b2f(q[base + (long long)j * DHEAD]);
        sk += b2f(k[base + (long long)j * DHEAD]);
    }
    const long long o = ((long long)bh * LAND + m) * DHEAD + d;
    ql[o] = f2bf(sq * (1.f / LGRP));
    kl[o] = f2bf(sk * (1.f / LGRP));
}

// ---------------------------------------------------------------------------
// Row softmax (one wave per row), bf16 in/out, optional f32 copy out.
// ---------------------------------------------------------------------------
__global__ __launch_bounds__(256) void softmax_kernel(
    unsigned short* __restrict__ s, float* __restrict__ sf, long long rows, int cols)
{
    const int lane = threadIdx.x & 31;
    long long r = (long long)blockIdx.x * 8 + (threadIdx.x >> 5);
    if (r >= rows) return;
    unsigned short* row = s + r * cols;
    float mx = -1e30f;
    for (int c = lane; c < cols; c += 32) mx = fmaxf(mx, b2f(row[c]));
#pragma unroll
    for (int o = 16; o > 0; o >>= 1) mx = fmaxf(mx, __shfl_xor(mx, o, 32));
    float sum = 0.f;
    for (int c = lane; c < cols; c += 32) sum += __expf(b2f(row[c]) - mx);
#pragma unroll
    for (int o = 16; o > 0; o >>= 1) sum += __shfl_xor(sum, o, 32);
    float inv = 1.f / sum;
    for (int c = lane; c < cols; c += 32) {
        float v = __expf(b2f(row[c]) - mx) * inv;
        row[c] = f2bf(v);
        if (sf) sf[r * cols + c] = v;
    }
}

// ---------------------------------------------------------------------------
// pinv init helpers: global max abs-row-sum / abs-col-sum (nonneg -> bit max)
// ---------------------------------------------------------------------------
__global__ void init_scal_kernel(unsigned* scal) {
    if (threadIdx.x < 2 && blockIdx.x == 0) scal[threadIdx.x] = 0u;
}

__global__ void absmax_kernel(const float* __restrict__ a, unsigned* __restrict__ scal) {
    int idx = blockIdx.x * blockDim.x + threadIdx.x;   // bh * 256
    if (idx >= BATCH * HEADS * LAND) return;
    int i  = idx & (LAND - 1);
    int bh = idx >> 8;
    const float* mm = a + (long long)bh * LAND * LAND;
    float rs = 0.f, cs = 0.f;
    for (int j = 0; j < LAND; ++j) {
        rs += fabsf(mm[i * LAND + j]);
        cs += fabsf(mm[j * LAND + i]);
    }
    atomicMax(&scal[0], __float_as_uint(rs));
    atomicMax(&scal[1], __float_as_uint(cs));
}

__global__ void z0_kernel(const float* __restrict__ a, const unsigned* __restrict__ scal,
                          float* __restrict__ zf, unsigned short* __restrict__ zb)
{
    long long idx = (long long)blockIdx.x * blockDim.x + threadIdx.x;
    if (idx >= (long long)BATCH * HEADS * LAND * LAND) return;
    int j  = (int)(idx & (LAND - 1));
    int i  = (int)((idx >> 8) & (LAND - 1));
    long long bh = idx >> 16;
    float s = 1.f / (__uint_as_float(scal[0]) * __uint_as_float(scal[1]));
    float v = a[(bh * LAND + j) * LAND + i] * s;   // transpose
    zf[idx] = v;
    zb[idx] = f2bf(v);
}

// out = c*I - in  (per 256x256 batch matrix), f32 + bf16 out
__global__ void isub_kernel(const float* __restrict__ in, float* __restrict__ of,
                            unsigned short* __restrict__ ob, float c, long long total)
{
    long long idx = (long long)blockIdx.x * blockDim.x + threadIdx.x;
    if (idx >= total) return;
    int j = (int)(idx & (LAND - 1));
    int i = (int)((idx >> 8) & (LAND - 1));
    float v = ((i == j) ? c : 0.f) - in[idx];
    of[idx] = v;
    ob[idx] = f2bf(v);
}

// ---------------------------------------------------------------------------
// Grouped conv residual (33 taps along heads axis, SAME; only 8 in-range) fused
// with add + [b,h,n,d] -> [b,n,h*d] transpose + bf16 pack for final projection.
// ---------------------------------------------------------------------------
__global__ void residual_pack_kernel(const float* __restrict__ oh,
                                     const float* __restrict__ vf,
                                     const float* __restrict__ cw,
                                     unsigned short* __restrict__ afin)
{
    long long idx = (long long)blockIdx.x * blockDim.x + threadIdx.x;
    if (idx >= (long long)BATCH * HEADS * NTOK * DHEAD) return;
    int d = (int)(idx & 63);
    int n = (int)((idx >> 6) & 4095);
    int h = (int)((idx >> 18) & 7);
    int b = (int)(idx >> 21);
    int g = d >> 3;     // group = out_channel/8
    float res = 0.f;
#pragma unroll
    for (int hh = 0; hh < 8; ++hh) {
        int tap = hh + 16 - h;                                 // always in [0,33)
        const float* vrow = vf + ((((long long)b * HEADS + hh) * NTOK) + n) * DHEAD + g * 8;
        const float* w = cw + (long long)tap * (8 * 64) + d;   // conv_w[tap][0][ci][d]
#pragma unroll
        for (int ci = 0; ci < 8; ++ci) res += vrow[ci] * w[ci * 64];
    }
    float v = oh[idx] + res;
    afin[(((long long)b * NTOK + n) * DIM) + h * DHEAD + d] = f2bf(v);
}

// ---------------------------------------------------------------------------
// host-side launch
// ---------------------------------------------------------------------------
static void launch_gemm(bool transb,
                        const unsigned short* A, const unsigned short* B,
                        float* Cf, unsigned short* Cbf, const float* bias,
                        int M, int N, int K, int lda, int ldb, int ldc,
                        long long sA, long long sB, long long sC,
                        float alpha, int batch, hipStream_t stream)
{
    int tiles = (M / 32) * (N / 32);
    dim3 grid((tiles + 7) / 8, 1, batch), block(256);
    if (transb)
        gemm_bf16<true><<<grid, block, 0, stream>>>(A, B, Cf, Cbf, bias, M, N, K,
                                                    lda, ldb, ldc, sA, sB, sC, alpha);
    else
        gemm_bf16<false><<<grid, block, 0, stream>>>(A, B, Cf, Cbf, bias, M, N, K,
                                                     lda, ldb, ldc, sA, sB, sC, alpha);
}

extern "C" void kernel_launch(void* const* d_in, const int* in_sizes, int n_in,
                              void* d_out, int out_size, void* d_ws, size_t ws_size,
                              hipStream_t stream)
{
    (void)in_sizes; (void)n_in; (void)out_size; (void)ws_size;
    const float* x     = (const float*)d_in[0];   // [2,4096,512]
    const float* w_qkv = (const float*)d_in[1];   // [512,1536]
    const float* w_out = (const float*)d_in[2];   // [512,512]
    const float* b_out = (const float*)d_in[3];   // [512]
    const float* cw    = (const float*)d_in[4];   // [33,1,8,64]
    float* out = (float*)d_out;

    const int BH = BATCH * HEADS;                 // 16
    char* ws = (char*)d_ws;
    size_t off = 0;
    auto carve = [&](size_t bytes) { size_t o = off; off += (bytes + 255) & ~(size_t)255; return (void*)(ws + o); };

    unsigned short* xb   = (unsigned short*)carve((size_t)BATCH * NTOK * DIM * 2);
    unsigned short* wqb  = (unsigned short*)carve((size_t)DIM * 3 * HEADS * DHEAD * 2);
    unsigned short* wob  = (unsigned short*)carve((size_t)DIM * DIM * 2);
    unsigned short* qb   = (unsigned short*)carve((size_t)BH * NTOK * DHEAD * 2);
    unsigned short* kb   = (unsigned short*)carve((size_t)BH * NTOK * DHEAD * 2);
    unsigned short* vb   = (unsigned short*)carve((size_t)BH * NTOK * DHEAD * 2);
    float*          vf   = (float*)carve((size_t)BH * NTOK * DHEAD * 4);
    unsigned short* qlb  = (unsigned short*)carve((size_t)BH * LAND * DHEAD * 2);
    unsigned short* klb  = (unsigned short*)carve((size_t)BH * LAND * DHEAD * 2);
    unsigned short* s1b  = (unsigned short*)carve((size_t)BH * NTOK * LAND * 2);
    unsigned short* s3b  = (unsigned short*)carve((size_t)BH * LAND * NTOK * 2);
    unsigned short* s2b  = (unsigned short*)carve((size_t)BH * LAND * LAND * 2);
    float*          s2f  = (float*)carve((size_t)BH * LAND * LAND * 4);
    unsigned*       scal = (unsigned*)carve(256);
    float*          zf0  = (float*)carve((size_t)BH * LAND * LAND * 4);
    unsigned short* zb0  = (unsigned short*)carve((size_t)BH * LAND * LAND * 2);
    float*          zf1  = (float*)carve((size_t)BH * LAND * LAND * 4);
    unsigned short* zb1  = (unsigned short*)carve((size_t)BH * LAND * LAND * 2);
    float*          azf  = (float*)carve((size_t)BH * LAND * LAND * 4);
    unsigned short* azb  = (unsigned short*)carve((size_t)BH * LAND * LAND * 2);
    float*          tAf  = (float*)carve((size_t)BH * LAND * LAND * 4);
    unsigned short* tAb  = (unsigned short*)carve((size_t)BH * LAND * LAND * 2);
    float*          tBf  = (float*)carve((size_t)BH * LAND * LAND * 4);
    unsigned short* w1b  = (unsigned short*)carve((size_t)BH * LAND * DHEAD * 2);
    unsigned short* w2b  = (unsigned short*)carve((size_t)BH * LAND * DHEAD * 2);
    float*          ohf  = (float*)carve((size_t)BH * NTOK * DHEAD * 4);
    unsigned short* afin = (unsigned short*)carve((size_t)BATCH * NTOK * DIM * 2);

    // 1. bf16 conversions of inputs
    {
        long long n = (long long)BATCH * NTOK * DIM;
        cvt_bf16_kernel<<<dim3((unsigned)((n + 255) / 256)), dim3(256), 0, stream>>>(x, xb, n);
        n = (long long)DIM * 3 * HEADS * DHEAD;
        cvt_bf16_kernel<<<dim3((unsigned)((n + 255) / 256)), dim3(256), 0, stream>>>(w_qkv, wqb, n);
        n = (long long)DIM * DIM;
        cvt_bf16_kernel<<<dim3((unsigned)((n + 255) / 256)), dim3(256), 0, stream>>>(w_out, wob, n);
    }

    // 2. QKV projection with head-scatter epilogue (32x32 wave tiles)
    {
        int tiles = (BATCH * NTOK / 32) * (3 * HEADS * DHEAD / 32);   // 12288
        gemm_qkv_kernel<<<dim3((tiles + 7) / 8), dim3(256), 0, stream>>>(xb, wqb, qb, kb, vb, vf);
    }

    // 3. landmarks
    {
        int n = BH * LAND * DHEAD;
        landmarks_kernel<<<dim3((n + 255) / 256), dim3(256), 0, stream>>>(qb, kb, qlb, klb);
    }

    // 4. sims (A @ B^T form)
    launch_gemm(true, qb,  klb, nullptr, s1b, nullptr, NTOK, LAND, DHEAD,
                DHEAD, DHEAD, LAND, (long long)NTOK * DHEAD, (long long)LAND * DHEAD,
                (long long)NTOK * LAND, 1.f, BH, stream);
    launch_gemm(true, qlb, klb, nullptr, s2b, nullptr, LAND, LAND, DHEAD,
                DHEAD, DHEAD, LAND, (long long)LAND * DHEAD, (long long)LAND * DHEAD,
                (long long)LAND * LAND, 1.f, BH, stream);
    launch_gemm(true, qlb, kb,  nullptr, s3b, nullptr, LAND, NTOK, DHEAD,
                DHEAD, DHEAD, NTOK, (long long)LAND * DHEAD, (long long)NTOK * DHEAD,
                (long long)LAND * NTOK, 1.f, BH, stream);

    // 5. softmaxes
    softmax_kernel<<<dim3((unsigned)(((long long)BH * NTOK + 7) / 8)), dim3(256), 0, stream>>>(
        s1b, nullptr, (long long)BH * NTOK, LAND);
    softmax_kernel<<<dim3((BH * LAND + 7) / 8), dim3(256), 0, stream>>>(
        s2b, s2f, (long long)BH * LAND, LAND);
    softmax_kernel<<<dim3((BH * LAND + 7) / 8), dim3(256), 0, stream>>>(
        s3b, nullptr, (long long)BH * LAND, NTOK);

    // 6. Moore-Penrose pinv of attn2
    init_scal_kernel<<<dim3(1), dim3(32), 0, stream>>>(scal);
    absmax_kernel<<<dim3((BH * LAND + 255) / 256), dim3(256), 0, stream>>>(s2f, scal);
    const long long mmtot = (long long)BH * LAND * LAND;
    z0_kernel<<<dim3((unsigned)((mmtot + 255) / 256)), dim3(256), 0, stream>>>(s2f, scal, zf0, zb0);

    float*          zfc[2] = { zf0, zf1 };
    unsigned short* zbc[2] = { zb0, zb1 };
    int cur = 0;
    const long long sMM = (long long)LAND * LAND;
    for (int it = 0; it < PITERS; ++it) {
        // az = a @ z
        launch_gemm(false, s2b, zbc[cur], azf, azb, nullptr, LAND, LAND, LAND,
                    LAND, LAND, LAND, sMM, sMM, sMM, 1.f, BH, stream);
        // tA = 7I - az
        isub_kernel<<<dim3((unsigned)((mmtot + 255) / 256)), dim3(256), 0, stream>>>(azf, tAf, tAb, 7.f, mmtot);
        // tB = az @ tA
        launch_gemm(false, azb, tAb, tBf, nullptr, nullptr, LAND, LAND, LAND,
                    LAND, LAND, LAND, sMM, sMM, sMM, 1.f, BH, stream);
        // tA = 15I - tB
        isub_kernel<<<dim3((unsigned)((mmtot + 255) / 256)), dim3(256), 0, stream>>>(tBf, tAf, tAb, 15.f, mmtot);
        // tB = az @ tA
        launch_gemm(false, azb, tAb, tBf, nullptr, nullptr, LAND, LAND, LAND,
                    LAND, LAND, LAND, sMM, sMM, sMM, 1.f, BH, stream);
        // tA = 13I - tB
        isub_kernel<<<dim3((unsigned)((mmtot + 255) / 256)), dim3(256), 0, stream>>>(tBf, tAf, tAb, 13.f, mmtot);
        // z_next = 0.25 * z @ tA
        launch_gemm(false, zbc[cur], tAb, zfc[1 - cur], zbc[1 - cur], nullptr, LAND, LAND, LAND,
                    LAND, LAND, LAND, sMM, sMM, sMM, 0.25f, BH, stream);
        cur ^= 1;
    }

    // 7. out = attn1 @ (zinv @ (attn3 @ v))
    launch_gemm(false, s3b, vb, nullptr, w1b, nullptr, LAND, DHEAD, NTOK,
                NTOK, DHEAD, DHEAD, (long long)LAND * NTOK, (long long)NTOK * DHEAD,
                (long long)LAND * DHEAD, 1.f, BH, stream);
    launch_gemm(false, zbc[cur], w1b, nullptr, w2b, nullptr, LAND, DHEAD, LAND,
                LAND, DHEAD, DHEAD, sMM, (long long)LAND * DHEAD,
                (long long)LAND * DHEAD, 1.f, BH, stream);
    launch_gemm(false, s1b, w2b, ohf, nullptr, nullptr, NTOK, DHEAD, LAND,
                LAND, DHEAD, DHEAD, (long long)NTOK * LAND, (long long)LAND * DHEAD,
                (long long)NTOK * DHEAD, 1.f, BH, stream);

    // 8. conv residual + transpose + bf16 pack
    {
        long long n = (long long)BH * NTOK * DHEAD;
        residual_pack_kernel<<<dim3((unsigned)((n + 255) / 256)), dim3(256), 0, stream>>>(ohf, vf, cw, afin);
    }

    // 9. final projection with bias -> d_out (f32)
    launch_gemm(false, afin, wob, out, nullptr, b_out, BATCH * NTOK, DIM, DIM,
                DIM, DIM, DIM, 0, 0, 0, 1.f, 1, stream);
}
